// GraphAutoEncoder_85959475462613
// MI455X (gfx1250) — matrix operand order repaired and checked by reading, compiled, not run on hardware
//
#include <hip/hip_runtime.h>
#include <hip/hip_bf16.h>

#define NUM_NODES 51681

typedef __attribute__((ext_vector_type(16))) __bf16       bf16x16;
typedef __attribute__((ext_vector_type(2)))  __bf16       bf16x2;
typedef __attribute__((ext_vector_type(8)))  float        f32x8;
typedef __attribute__((ext_vector_type(4)))  unsigned int u32x4;
typedef __attribute__((ext_vector_type(8)))  unsigned int u32x8;

__device__ __forceinline__ unsigned int pack_bf16(float lo, float hi) {
    bf16x2 p;
    p[0] = (__bf16)lo;
    p[1] = (__bf16)hi;
    return __builtin_bit_cast(unsigned int, p);
}

constexpr int ilog2(int x) { int s = 0; while ((1 << s) < x) ++s; return s; }

// ---------------------------------------------------------------------------
// Degree / normalization (sparse part: 2048 edges, trivial)
// ---------------------------------------------------------------------------
__global__ void deg_init_kernel(float* __restrict__ deg, int n) {
    int i = blockIdx.x * blockDim.x + threadIdx.x;
    if (i < n) deg[i] = 1.0f;   // self-loop weight
}

__global__ void deg_edge_kernel(float* __restrict__ deg,
                                const int* __restrict__ dst,
                                const float* __restrict__ w, int e) {
    int i = blockIdx.x * blockDim.x + threadIdx.x;
    if (i < e) atomicAdd(&deg[dst[i]], w[i]);
}

__global__ void dinv_kernel(float* __restrict__ dinv, int n) {
    int i = blockIdx.x * blockDim.x + threadIdx.x;
    if (i < n) {
        float d = dinv[i];
        dinv[i] = (d > 0.0f) ? __frsqrt_rn(d) : 0.0f;
    }
}

// coef'[e] = w[e] * dinv[dst[e]] / dinv[src[e]]  (applied to the ALREADY
// self-scaled output:  coef' * dinv[src]^2 * xw[src] == dinv[src]*w*dinv[dst]*xw[src])
__global__ void coef_kernel(float* __restrict__ coef,
                            const float* __restrict__ dinv,
                            const int* __restrict__ src,
                            const int* __restrict__ dst,
                            const float* __restrict__ w, int e) {
    int i = blockIdx.x * blockDim.x + threadIdx.x;
    if (i < e) {
        float ds = dinv[src[i]];
        coef[i] = (ds > 0.0f) ? (w[i] * dinv[dst[i]] / ds) : 0.0f;
    }
}

// ---------------------------------------------------------------------------
// Fused WMMA bf16 GEMM for GCN:
//   Y[row, col] = (dinv[row]^2 * X[row,:]) @ W[:,col] + bias[col]
// The self-loop scale dinv^2 is folded into A staging (row scaling commutes
// with the GEMM), so the epilogue is a pure  acc + bias  store.
// Block = 128 threads (4 waves). Block tile: 16 rows x (64*CT) cols.
// A/B staged in LDS as packed bf16 K-pairs in fragment order so every
// fragment is ds_load_b128. One v_wmma_f32_16x16x32_bf16 per K-slice x tile.
// ---------------------------------------------------------------------------
template <int K, int FOUT, int CT, bool NT>
__global__ void gemm_fused_kernel(const float* __restrict__ X,
                                  const float* __restrict__ W,
                                  const float* __restrict__ bias,
                                  const float* __restrict__ dinv,
                                  float* __restrict__ Y,
                                  int nrows) {
    constexpr int KP    = K / 2;                  // u32 K-pairs per row
    constexpr int SA    = (KP + 4 + 3) & ~3;      // padded LDS stride (x4 aligned)
    constexpr int BCOLS = 64 * CT;
    constexpr bool COLS_EXACT = (FOUT % BCOLS) == 0;

    __shared__ __align__(16) unsigned int Asw[16 * SA];     // [m][kp]
    __shared__ __align__(16) unsigned int Bsw[BCOLS * SA];  // [n][kp]

    const int tid  = threadIdx.x;
    const int wave = tid >> 5;
    const int lane = tid & 31;

    const int row0 = blockIdx.y * 16;
    const int col0 = blockIdx.x * BCOLS;

    // ---- stage A (16 x K) as packed pairs, pre-scaled by dinv^2 ----
    for (int idx = tid; idx < 16 * KP; idx += 128) {
        int r = idx / KP, jj = idx % KP;
        int g = row0 + r;
        float v0 = 0.f, v1 = 0.f;
        if (g < nrows) {
            float d = dinv[g];
            float s = d * d;
            const float* p = X + (size_t)g * K + 2 * jj;
            v0 = p[0] * s; v1 = p[1] * s;
        }
        Asw[r * SA + jj] = pack_bf16(v0, v1);
    }
    // ---- stage B (K x BCOLS) as packed pairs ----
    for (int idx = tid; idx < BCOLS * KP; idx += 128) {
        int jj = idx / BCOLS, n = idx % BCOLS;   // n fast => coalesced W reads
        int gc = col0 + n;
        float v0 = 0.f, v1 = 0.f;
        if (COLS_EXACT || gc < FOUT) {
            v0 = W[(size_t)(2 * jj) * FOUT + gc];
            v1 = W[(size_t)(2 * jj + 1) * FOUT + gc];
        }
        Bsw[n * SA + jj] = pack_bf16(v0, v1);
    }
    __syncthreads();

    const int m    = lane & 15;
    const int half = lane >> 4;

    f32x8 acc[CT];
    #pragma unroll
    for (int ct = 0; ct < CT; ++ct) acc[ct] = (f32x8){0.f,0.f,0.f,0.f,0.f,0.f,0.f,0.f};

    #pragma unroll
    for (int kp0 = 0; kp0 < KP; kp0 += 16) {     // one 16x16x32 slice per iter
        // A fragment (ISA 16-bit A 16x32 layout), two b128 LDS loads
        u32x4 a0 = *(const u32x4*)&Asw[m * SA + kp0 + half * 4];
        u32x4 a1 = *(const u32x4*)&Asw[m * SA + kp0 + 8 + half * 4];
        u32x8 au;
        #pragma unroll
        for (int j = 0; j < 4; ++j) { au[j] = a0[j]; au[4 + j] = a1[j]; }
        bf16x16 a = __builtin_bit_cast(bf16x16, au);

        #pragma unroll
        for (int ct = 0; ct < CT; ++ct) {
            int n = ct * 64 + (wave << 4) + m;
            // B fragment (ISA 16-bit B 32x16 layout), two b128 LDS loads
            u32x4 b0 = *(const u32x4*)&Bsw[n * SA + kp0 + half * 4];
            u32x4 b1 = *(const u32x4*)&Bsw[n * SA + kp0 + 8 + half * 4];
            u32x8 bu;
            #pragma unroll
            for (int j = 0; j < 4; ++j) { bu[j] = b0[j]; bu[4 + j] = b1[j]; }
            bf16x16 b = __builtin_bit_cast(bf16x16, bu);

            acc[ct] = __builtin_amdgcn_wmma_f32_16x16x32_bf16(
                false, a, false, b, (short)0, acc[ct], false, false);
        }
    }

    // ---- epilogue: acc + bias; uniform branch between interior / edge ----
    const int rbase = row0 + (half << 3);        // first of this lane's 8 rows
    if (row0 + 16 <= nrows) {
        // interior: straight-line stores, compile-time r*FOUT offsets
        #pragma unroll
        for (int ct = 0; ct < CT; ++ct) {
            int col = col0 + ct * 64 + (wave << 4) + m;
            if (COLS_EXACT || col < FOUT) {
                float bv = bias[col];
                float* p = Y + (size_t)rbase * FOUT + col;
                #pragma unroll
                for (int r = 0; r < 8; ++r) {
                    float val = acc[ct][r] + bv;
                    if (NT) __builtin_nontemporal_store(val, p + (size_t)r * FOUT);
                    else    p[(size_t)r * FOUT] = val;
                }
            }
        }
    } else {
        #pragma unroll
        for (int ct = 0; ct < CT; ++ct) {
            int col = col0 + ct * 64 + (wave << 4) + m;
            if (COLS_EXACT || col < FOUT) {
                float bv = bias[col];
                #pragma unroll
                for (int r = 0; r < 8; ++r) {
                    int row = rbase + r;
                    if (row < nrows)
                        Y[(size_t)row * FOUT + col] = acc[ct][r] + bv;
                }
            }
        }
    }
}

// ---------------------------------------------------------------------------
// Edge message gather from the self-scaled output:
//   msgs[e][f] = coef'[e] * (out[src[e]][f] - bias[f])
// ---------------------------------------------------------------------------
template <int FOUT>
__global__ void gather_kernel(const float* __restrict__ out,
                              const float* __restrict__ bias,
                              const float* __restrict__ coef,
                              const int* __restrict__ src,
                              float* __restrict__ msgs, int e) {
    constexpr int SH = ilog2(FOUT);
    long long idx = (long long)blockIdx.x * blockDim.x + threadIdx.x;
    if (idx < ((long long)e << SH)) {
        int ei = (int)(idx >> SH);
        int f  = (int)(idx & (FOUT - 1));
        msgs[idx] = coef[ei] * (out[((size_t)src[ei] << SH) + f] - bias[f]);
    }
}

template <int FOUT>
__global__ void scatter_kernel(const float* __restrict__ msgs,
                               const int* __restrict__ dst,
                               float* __restrict__ out, int e) {
    constexpr int SH = ilog2(FOUT);
    long long idx = (long long)blockIdx.x * blockDim.x + threadIdx.x;
    if (idx < ((long long)e << SH)) {
        int ei = (int)(idx >> SH);
        int f  = (int)(idx & (FOUT - 1));
        atomicAdd(&out[((size_t)dst[ei] << SH) + f], msgs[idx]);
    }
}

__global__ void relu_kernel(float* __restrict__ out, long long total) {
    long long idx = (long long)blockIdx.x * blockDim.x + threadIdx.x;
    if (idx < total) out[idx] = fmaxf(out[idx], 0.0f);
}

// ---------------------------------------------------------------------------
// Driver
// ---------------------------------------------------------------------------
static inline int cdiv(long long a, int b) { return (int)((a + b - 1) / b); }

template <int K, int FOUT, int CT, bool NT>
static void run_layer(const float* x, const float* W, const float* bias,
                      float* out, float* msgs,
                      const float* dinv, const float* coef,
                      const int* src, const int* dst,
                      int n, int E, bool relu, hipStream_t stream) {
    dim3 ggrid((FOUT + 64 * CT - 1) / (64 * CT), (n + 15) / 16);
    gemm_fused_kernel<K, FOUT, CT, NT><<<ggrid, 128, 0, stream>>>(x, W, bias, dinv, out, n);

    long long etotal = (long long)E * FOUT;
    gather_kernel<FOUT><<<cdiv(etotal, 256), 256, 0, stream>>>(out, bias, coef, src, msgs, E);
    scatter_kernel<FOUT><<<cdiv(etotal, 256), 256, 0, stream>>>(msgs, dst, out, E);

    if (relu) {
        long long ntotal = (long long)n * FOUT;
        relu_kernel<<<cdiv(ntotal, 256), 256, 0, stream>>>(out, ntotal);
    }
}

extern "C" void kernel_launch(void* const* d_in, const int* in_sizes, int n_in,
                              void* d_out, int out_size, void* d_ws, size_t ws_size,
                              hipStream_t stream) {
    (void)n_in; (void)out_size; (void)ws_size;
    const int N = NUM_NODES;
    const int E = in_sizes[1];   // 2048

    const int*   edge_index = (const int*)d_in[0];
    const int*   src = edge_index;
    const int*   dst = edge_index + E;
    const float* ew      = (const float*)d_in[1];
    const float* emb     = (const float*)d_in[2];
    const float* enc_W1  = (const float*)d_in[3];
    const float* enc_b1  = (const float*)d_in[4];
    const float* enc_W2  = (const float*)d_in[5];
    const float* enc_b2  = (const float*)d_in[6];
    const float* dec_W1  = (const float*)d_in[7];
    const float* dec_b1  = (const float*)d_in[8];
    const float* dec_W2  = (const float*)d_in[9];
    const float* dec_b2  = (const float*)d_in[10];

    float* recon = (float*)d_out;                       // [N, 2048]
    float* z     = (float*)d_out + (size_t)N * 2048;    // [N, 32]

    // Workspace layout (floats)
    float* ws   = (float*)d_ws;
    float* dinv = ws;                                   // N (deg first, then rsqrt)
    float* coef = dinv + 51712;                         // E
    float* bufB = coef + 2048;                          // N*64  (x1)
    float* bufC = bufB + (size_t)N * 64;                // N*64  (h)
    float* msgs = bufC + (size_t)N * 64;                // E*2048 max

    // Normalization precompute
    deg_init_kernel<<<cdiv(N, 256), 256, 0, stream>>>(dinv, N);
    deg_edge_kernel<<<cdiv(E, 256), 256, 0, stream>>>(dinv, dst, ew, E);
    dinv_kernel<<<cdiv(N, 256), 256, 0, stream>>>(dinv, N);
    coef_kernel<<<cdiv(E, 256), 256, 0, stream>>>(coef, dinv, src, dst, ew, E);

    // Encoder
    run_layer<32,   64, 1, false>(emb,  enc_W1, enc_b1, bufB,  msgs, dinv, coef, src, dst, N, E, true,  stream);
    run_layer<64,   32, 1, false>(bufB, enc_W2, enc_b2, z,     msgs, dinv, coef, src, dst, N, E, true,  stream);
    // Decoder
    run_layer<32,   64, 1, false>(z,    dec_W1, dec_b1, bufC,  msgs, dinv, coef, src, dst, N, E, true,  stream);
    run_layer<64, 2048, 4, true >(bufC, dec_W2, dec_b2, recon, msgs, dinv, coef, src, dst, N, E, false, stream);
}
